// TransformerBlock_87703232185079
// MI455X (gfx1250) — matrix operand order repaired and checked by reading, not compiled, not run on hardware
//
#include <hip/hip_runtime.h>

typedef unsigned short u16;
typedef __attribute__((ext_vector_type(16))) __bf16 v16bf;
typedef __attribute__((ext_vector_type(8)))  __bf16 v8bf;
typedef __attribute__((ext_vector_type(8)))  float  v8f;
typedef __attribute__((ext_vector_type(8)))  u16    v8u;
typedef __attribute__((ext_vector_type(4)))  unsigned u32x4;
typedef __attribute__((ext_vector_type(8)))  int      i32x8;
typedef __attribute__((ext_vector_type(4)))  int      i32x4;

static constexpr int Bc   = 16;
static constexpr int Sc   = 512;
static constexpr int Dc   = 1024;
static constexpr int Hc   = 16;
static constexpr int DKc  = 64;
static constexpr int DFFc = 4096;
static constexpr int MT   = Bc * Sc;      // 8192 token rows
static constexpr float EPSc = 1e-5f;

// Tensor Data Mover availability (ROCm7.2 clang-22 = 5-arg, clang-23 = 6-arg)
#if defined(__has_builtin)
# if __has_builtin(__builtin_amdgcn_tensor_load_to_lds) && \
     __has_builtin(__builtin_amdgcn_s_wait_tensorcnt)
#  define TDM_OK 1
# endif
#endif
#ifndef TDM_OK
# define TDM_OK 0
#endif

// ---------------- bf16 <-> f32 (storage = ushort, RNE convert) ----------------
__device__ __forceinline__ u16 f2bf(float f) {
  union { float f; unsigned u; } c; c.f = f;
  unsigned r = c.u + 0x7FFFu + ((c.u >> 16) & 1u);
  return (u16)(r >> 16);
}
__device__ __forceinline__ v8f vz8() {
  v8f z = {0.f,0.f,0.f,0.f,0.f,0.f,0.f,0.f}; return z;
}
__device__ __forceinline__ float redmax16(float v) {
  v = fmaxf(v, __shfl_xor(v, 1, 16));
  v = fmaxf(v, __shfl_xor(v, 2, 16));
  v = fmaxf(v, __shfl_xor(v, 4, 16));
  v = fmaxf(v, __shfl_xor(v, 8, 16));
  return v;
}
__device__ __forceinline__ float redsum16(float v) {
  v += __shfl_xor(v, 1, 16);
  v += __shfl_xor(v, 2, 16);
  v += __shfl_xor(v, 4, 16);
  v += __shfl_xor(v, 8, 16);
  return v;
}
__device__ __forceinline__ unsigned lds_off(const void* p) {
  // generic shared-aperture address: low 32 bits are the LDS byte offset
  return (unsigned)(size_t)p;
}

// ---- TDM: 2D tile global -> LDS, row-major, optional LDS row padding --------
// D# group0: count=1 | lds_addr | global_addr(57b) | type=2
// D# group1: data_size=2B | pad ctl | tensor dims/tile dims | dim0 stride
__device__ __forceinline__ void tdm_load_tile_2d(
    const void* gsrc, unsigned ldsOff,
    unsigned tileD0, unsigned tileD1, unsigned strideD0,
    unsigned padIntervalCode, unsigned padAmountCode) {
#if TDM_OK
  unsigned long long ga = (unsigned long long)(size_t)gsrc;
  u32x4 g0;
  g0[0] = 1u;                                           // count=1, user mode
  g0[1] = ldsOff;                                       // LDS byte address
  g0[2] = (unsigned)ga;                                 // global addr lo
  g0[3] = (unsigned)((ga >> 32) & 0x01FFFFFFull) | (2u << 30);  // hi25 | type=2
  i32x8 g1;
  g1[0] = (int)((1u << 16) |                            // data_size = 2 bytes
                (1u << 20) |                            // pad_enable
                (padIntervalCode << 22) |
                (padAmountCode << 25));
  g1[1] = (int)((tileD0 & 0xFFFFu) << 16);              // tensor_dim0 lo16 (== tile)
  g1[2] = (int)(((tileD0 >> 16) & 0xFFFFu) | ((tileD1 & 0xFFFFu) << 16)); // tensor_dim1
  g1[3] = (int)(((tileD1 >> 16) & 0xFFFFu) | ((tileD0 & 0xFFFFu) << 16)); // tile_dim0
  g1[4] = (int)(tileD1 & 0xFFFFu);                      // tile_dim1 | tile_dim2=0
  g1[5] = (int)strideD0;                                // tensor_dim0_stride lo32
  g1[6] = 0;                                            // stride hi | dim1_stride lo
  g1[7] = 0;                                            // dim1_stride hi
  i32x4 z4 = {0, 0, 0, 0};
# if __clang_major__ >= 23
  i32x8 z8 = {0, 0, 0, 0, 0, 0, 0, 0};
  __builtin_amdgcn_tensor_load_to_lds(g0, g1, z4, z4, z8, 0);
# else
  __builtin_amdgcn_tensor_load_to_lds(g0, g1, z4, z4, 0);
# endif
#else
  (void)gsrc; (void)ldsOff; (void)tileD0; (void)tileD1;
  (void)strideD0; (void)padIntervalCode; (void)padAmountCode;
#endif
}

// ---------------- weight abs-max (for fake-quant scale) ----------------
__global__ void init_amax_kernel(unsigned* a) {
  if (threadIdx.x < 3) a[threadIdx.x] = 0u;
}

__global__ __launch_bounds__(256) void absmax_kernel(const float* __restrict__ w,
                                                     int n, unsigned* __restrict__ out) {
  float m = 0.f;
  for (int i = blockIdx.x * blockDim.x + threadIdx.x; i < n; i += gridDim.x * blockDim.x)
    m = fmaxf(m, fabsf(w[i]));
#pragma unroll
  for (int s = 16; s >= 1; s >>= 1) m = fmaxf(m, __shfl_xor(m, s, 32));
  __shared__ float sm[8];
  if ((threadIdx.x & 31) == 0) sm[threadIdx.x >> 5] = m;
  __syncthreads();
  if (threadIdx.x == 0) {
    float t = sm[0];
#pragma unroll
    for (int i = 1; i < 8; ++i) t = fmaxf(t, sm[i]);
    atomicMax(out, __float_as_uint(t));
  }
}

// fake-quant int8 symmetric, bf16 convert, TRANSPOSE: w[K][N] -> o[N][K]
__global__ __launch_bounds__(256) void quant_bf16_t_kernel(
    const float* __restrict__ w, u16* __restrict__ o, int Kd, int Nd,
    const unsigned* __restrict__ amax) {
  long long i = (long long)blockIdx.x * 256 + threadIdx.x;
  if (i >= (long long)Kd * Nd) return;
  int n = (int)(i / Kd), k = (int)(i % Kd);
  float scale = __uint_as_float(*amax) * (1.0f / 127.0f);
  float inv = (scale > 0.f) ? (1.f / scale) : 0.f;
  o[i] = f2bf(roundf(w[(size_t)k * Nd + n] * inv) * scale);
}

// bf16 convert with transpose: w[K][N] f32 -> o[N][K] bf16
__global__ __launch_bounds__(256) void cvt_bf16_t_kernel(
    const float* __restrict__ w, u16* __restrict__ o, int Kd, int Nd) {
  long long i = (long long)blockIdx.x * 256 + threadIdx.x;
  if (i >= (long long)Kd * Nd) return;
  int n = (int)(i / Kd), k = (int)(i % Kd);
  o[i] = f2bf(w[(size_t)k * Nd + n]);
}

// ---------------- LayerNorm (f32 in) -> bf16 out ----------------
__global__ __launch_bounds__(256) void ln_bf16_kernel(const float* __restrict__ x,
                                                      const float* __restrict__ g,
                                                      const float* __restrict__ b,
                                                      u16* __restrict__ out) {
  const int row = blockIdx.x;
  const float* xr = x + (size_t)row * Dc;
  float vals[4];
  float s = 0.f, s2 = 0.f;
#pragma unroll
  for (int i = 0; i < 4; ++i) {
    float v = xr[threadIdx.x + i * 256];
    vals[i] = v; s += v; s2 += v * v;
  }
#pragma unroll
  for (int m = 16; m >= 1; m >>= 1) { s += __shfl_xor(s, m, 32); s2 += __shfl_xor(s2, m, 32); }
  __shared__ float w1s[8], w2s[8];
  if ((threadIdx.x & 31) == 0) { w1s[threadIdx.x >> 5] = s; w2s[threadIdx.x >> 5] = s2; }
  __syncthreads();
  float ts = 0.f, ts2 = 0.f;
#pragma unroll
  for (int i = 0; i < 8; ++i) { ts += w1s[i]; ts2 += w2s[i]; }
  const float mu  = ts * (1.0f / Dc);
  const float var = ts2 * (1.0f / Dc) - mu * mu;
  const float rs  = rsqrtf(var + EPSc);
#pragma unroll
  for (int i = 0; i < 4; ++i) {
    int c = threadIdx.x + i * 256;
    out[(size_t)row * Dc + c] = f2bf((vals[i] - mu) * rs * g[c] + b[c]);
  }
}

// ---------------- bf16 WMMA GEMM: C = A(MxK) * Wt(NxK)^T + bias [+epilogue] ---
// Weights are PRE-TRANSPOSED: Wt[N][K]. Both tiles are row-major [128][32]
// block copies -> staged by the Tensor Data Mover (double-buffered) when
// available, else plain vectorized copies.
// EP = 0: out bf16 = acc+bias
// EP = 1: out bf16 = gelu_exact(acc+bias)
// EP = 2: out f32  = res + acc + bias
static constexpr int BMt = 128, BNt = 128, BKt = 32, PAD = 8;
// LDS row: 32 elts (64B = 16 dwords) + 8 elts pad (16B = 4 dwords)
static constexpr unsigned PAD_INT_GEMM = 3;  // pad after 1<<3 = 8 x 8B = 64B
static constexpr unsigned PAD_AMT_GEMM = 3;  // 4 dwords = 16B

template <int EP>
__global__ __launch_bounds__(256) void gemm_bf16_kernel(
    const u16* __restrict__ A, const u16* __restrict__ Wt,
    const float* __restrict__ bias, const float* __restrict__ res,
    float* __restrict__ outF, u16* __restrict__ outH, int N, int K) {
  __shared__ u16 sA[2][BMt][BKt + PAD];
  __shared__ u16 sB[2][BNt][BKt + PAD];   // [n][k]

  const int tid  = threadIdx.x;
  const int lane = tid & 31;
  const int wave = tid >> 5;
  const int wm   = wave >> 1;   // 0..3 : 32-row strip
  const int wn   = wave & 1;    // 0..1 : 64-col strip
  const int bm0  = blockIdx.y * BMt;
  const int bn0  = blockIdx.x * BNt;

  v8f acc[2][4];
#pragma unroll
  for (int i = 0; i < 2; ++i)
#pragma unroll
    for (int j = 0; j < 4; ++j) acc[i][j] = vz8();

  const int srow = tid >> 1;          // 0..127
  const int scol = (tid & 1) * 16;    // 0 / 16
  const int fr   = lane & 15;
  const int lh   = lane >> 4;

  auto stage = [&](int buf, int k0) {
#if TDM_OK
    if (wave == 0) {
      tdm_load_tile_2d(A + (size_t)bm0 * K + k0, lds_off(&sA[buf][0][0]),
                       BKt, BMt, (unsigned)K, PAD_INT_GEMM, PAD_AMT_GEMM);
      tdm_load_tile_2d(Wt + (size_t)bn0 * K + k0, lds_off(&sB[buf][0][0]),
                       BKt, BNt, (unsigned)K, PAD_INT_GEMM, PAD_AMT_GEMM);
    }
#else
    const u16* ga = A + (size_t)(bm0 + srow) * K + k0 + scol;
    *(v8u*)&sA[buf][srow][scol]     = *(const v8u*)ga;
    *(v8u*)&sA[buf][srow][scol + 8] = *(const v8u*)(ga + 8);
    const u16* gb = Wt + (size_t)(bn0 + srow) * K + k0 + scol;
    *(v8u*)&sB[buf][srow][scol]     = *(const v8u*)gb;
    *(v8u*)&sB[buf][srow][scol + 8] = *(const v8u*)(gb + 8);
    __builtin_prefetch(ga + BKt, 0, 0);
    __builtin_prefetch(gb + BKt, 0, 0);
#endif
  };

  const int nk = K / BKt;
  stage(0, 0);
  for (int kt = 0; kt < nk; ++kt) {
    const int cur = kt & 1;
    const bool pre = (kt + 1) < nk;
    if (pre) stage(cur ^ 1, (kt + 1) * BKt);
#if TDM_OK
    if (wave == 0) {
      if (pre) __builtin_amdgcn_s_wait_tensorcnt((short)2);  // current pair done
      else     __builtin_amdgcn_s_wait_tensorcnt((short)0);
    }
#endif
    __syncthreads();

    const int kbA = lh * 8;   // A-frag: lanes 0-15 K=0..7|16..23, lanes 16-31 K=8..15|24..31
    const int kbB = lh * 8;   // B-frag: lanes 0-15 K=0..7|16..23, lanes 16-31 K=8..15|24..31
#pragma unroll
    for (int im = 0; im < 2; ++im) {
      v16bf af;
      const u16* pa = &sA[cur][wm * 32 + im * 16 + fr][kbA];
      ((v8bf*)&af)[0] = *(const v8bf*)pa;
      ((v8bf*)&af)[1] = *(const v8bf*)(pa + 16);
#pragma unroll
      for (int in = 0; in < 4; ++in) {
        v16bf bfv;
        const u16* pb = &sB[cur][wn * 64 + in * 16 + fr][kbB];
        ((v8bf*)&bfv)[0] = *(const v8bf*)pb;
        ((v8bf*)&bfv)[1] = *(const v8bf*)(pb + 16);
        acc[im][in] = __builtin_amdgcn_wmma_f32_16x16x32_bf16(
            false, af, false, bfv, (short)0, acc[im][in], false, false);
      }
    }
    __syncthreads();
  }

  // epilogue: C-layout lane (r, fr): row = r + lh*8, col = fr
#pragma unroll
  for (int im = 0; im < 2; ++im) {
#pragma unroll
    for (int in = 0; in < 4; ++in) {
      const int col = bn0 + wn * 64 + in * 16 + fr;
      const float bv = bias[col];
#pragma unroll
      for (int r = 0; r < 8; ++r) {
        const int row = bm0 + wm * 32 + im * 16 + lh * 8 + r;
        const size_t idx = (size_t)row * N + col;
        float v = acc[im][in][r] + bv;
        if (EP == 0) {
          outH[idx] = f2bf(v);
        } else if (EP == 1) {
          float gv = 0.5f * v * (1.f + erff(v * 0.70710678118654752f));
          outH[idx] = f2bf(gv);
        } else {
          outF[idx] = res[idx] + v;
        }
      }
    }
  }
}

// ---------------- flash attention: bf16 WMMA QK^T + PV, online softmax --------
// grid: (B*H*(S/128)) blocks, 256 threads (8 waves, 16 q-rows each)
// K tile (64x64, row-major) staged by TDM; V tile needs transpose -> manual.
__global__ __launch_bounds__(256) void attention_kernel(
    const u16* __restrict__ Q, const u16* __restrict__ Kk, const u16* __restrict__ V,
    const int* __restrict__ mask, u16* __restrict__ out) {
  __shared__ u16 sK[64][DKc + PAD];       // [kv][dk]
  __shared__ u16 sVt[DKc][64 + PAD];      // [dk][kv]
  __shared__ u16 sP[8][16][64 + PAD];     // per-wave P staging

  const int bid = blockIdx.x;
  const int qb  = bid & 3;
  const int bh  = bid >> 2;
  const int h   = bh & (Hc - 1);
  const int b   = bh >> 4;

  const int tid  = threadIdx.x;
  const int lane = tid & 31;
  const int wave = tid >> 5;
  const int fr   = lane & 15;
  const int lh   = lane >> 4;

  const int q0 = qb * 128 + wave * 16;                       // q-row base (in S)
  const size_t base = ((size_t)b * Sc) * Dc + h * DKc;       // [b][s][h*DK]

  // Q fragments (16 x 64 -> 2 A-frags of 16x32), loaded once
  v16bf qf[2];
  {
    const u16* qrow = Q + base + (size_t)(q0 + fr) * Dc;
#pragma unroll
    for (int kf = 0; kf < 2; ++kf) {
      const u16* p = qrow + kf * 32 + lh * 8;
      ((v8bf*)&qf[kf])[0] = *(const v8bf*)p;
      ((v8bf*)&qf[kf])[1] = *(const v8bf*)(p + 16);
    }
  }

  v8f o[4];
#pragma unroll
  for (int i = 0; i < 4; ++i) o[i] = vz8();
  float rm[8], rl[8];
#pragma unroll
  for (int r = 0; r < 8; ++r) { rm[r] = -1e30f; rl[r] = 0.f; }

  const int krow = tid >> 2;          // 0..63
  const int kcol = (tid & 3) * 16;    // 0..48

  for (int kv0 = 0; kv0 < Sc; kv0 += 64) {
    // stage K tile: DMA when available (row = 64 elts = 128B = 16x8B -> code 4)
#if TDM_OK
    if (wave == 0) {
      tdm_load_tile_2d(Kk + base + (size_t)kv0 * Dc, lds_off(&sK[0][0]),
                       DKc, 64, Dc, 4u, 3u);
    }
#else
    {
      const u16* gk = Kk + base + (size_t)(kv0 + krow) * Dc + kcol;
      *(v8u*)&sK[krow][kcol]     = *(const v8u*)gk;
      *(v8u*)&sK[krow][kcol + 8] = *(const v8u*)(gk + 8);
    }
#endif
    // stage V transposed (manual)
    {
      const u16* gv = V + base + (size_t)(kv0 + krow) * Dc + kcol;
      v8u t0 = *(const v8u*)gv;
      v8u t1 = *(const v8u*)(gv + 8);
#pragma unroll
      for (int j = 0; j < 8; ++j) sVt[kcol + j][krow]     = t0[j];
#pragma unroll
      for (int j = 0; j < 8; ++j) sVt[kcol + 8 + j][krow] = t1[j];
    }
#if TDM_OK
    if (wave == 0) __builtin_amdgcn_s_wait_tensorcnt((short)0);
#endif
    __syncthreads();

    // scores: 16 x 64 = 4 accum frags; contraction over DK=64 (2 WMMA steps)
    v8f sc[4];
#pragma unroll
    for (int nf = 0; nf < 4; ++nf) {
      sc[nf] = vz8();
#pragma unroll
      for (int kf = 0; kf < 2; ++kf) {
        v16bf bfv;   // B[k][n] = K[kv0+nf*16+n][kf*32+k] -> sK row contiguous
        const u16* pk = &sK[nf * 16 + fr][kf * 32 + lh * 8];
        ((v8bf*)&bfv)[0] = *(const v8bf*)pk;
        ((v8bf*)&bfv)[1] = *(const v8bf*)(pk + 16);
        sc[nf] = __builtin_amdgcn_wmma_f32_16x16x32_bf16(
            false, qf[kf], false, bfv, (short)0, sc[nf], false, false);
      }
    }

    // scale + mask
#pragma unroll
    for (int nf = 0; nf < 4; ++nf) {
      const int kg = kv0 + nf * 16 + fr;
#pragma unroll
      for (int r = 0; r < 8; ++r) {
        const int qr = q0 + lh * 8 + r;
        float v = sc[nf][r] * 0.125f;  // 1/sqrt(64)
        int mk = mask[((size_t)b * Sc + qr) * Sc + kg];
        sc[nf][r] = (mk == 0) ? -1e9f : v;
      }
    }

    // online softmax (per-register row stats align with lane halves)
    float alpha[8];
#pragma unroll
    for (int r = 0; r < 8; ++r) {
      float m = fmaxf(fmaxf(sc[0][r], sc[1][r]), fmaxf(sc[2][r], sc[3][r]));
      m = redmax16(m);
      float nm = fmaxf(rm[r], m);
      alpha[r] = expf(rm[r] - nm);
      rm[r] = nm;
      float ps = 0.f;
#pragma unroll
      for (int nf = 0; nf < 4; ++nf) {
        float p = expf(sc[nf][r] - nm);
        sc[nf][r] = p;
        ps += p;
      }
      ps = redsum16(ps);
      rl[r] = rl[r] * alpha[r] + ps;
#pragma unroll
      for (int nf = 0; nf < 4; ++nf) o[nf][r] *= alpha[r];
    }

    // P: C-layout -> LDS -> A-layout (per-wave private region)
#pragma unroll
    for (int nf = 0; nf < 4; ++nf)
#pragma unroll
      for (int r = 0; r < 8; ++r)
        sP[wave][lh * 8 + r][nf * 16 + fr] = f2bf(sc[nf][r]);

    v16bf pf[2];
#pragma unroll
    for (int kf = 0; kf < 2; ++kf) {
      const u16* pp = &sP[wave][fr][kf * 32 + lh * 8];
      ((v8bf*)&pf[kf])[0] = *(const v8bf*)pp;
      ((v8bf*)&pf[kf])[1] = *(const v8bf*)(pp + 16);
    }

    // O += P(16x64) * V(64x64)
#pragma unroll
    for (int nf = 0; nf < 4; ++nf) {
#pragma unroll
      for (int kf = 0; kf < 2; ++kf) {
        v16bf vb;   // Bt[n][k] = V[kv0+k][nf*16+n] -> sVt row contiguous
        const u16* pv = &sVt[nf * 16 + fr][kf * 32 + lh * 8];
        ((v8bf*)&vb)[0] = *(const v8bf*)pv;
        ((v8bf*)&vb)[1] = *(const v8bf*)(pv + 16);
        o[nf] = __builtin_amdgcn_wmma_f32_16x16x32_bf16(
            false, pf[kf], false, vb, (short)0, o[nf], false, false);
      }
    }
    __syncthreads();
  }

  // normalize + write attn in (B,S,D) layout
#pragma unroll
  for (int nf = 0; nf < 4; ++nf) {
#pragma unroll
    for (int r = 0; r < 8; ++r) {
      const int qr = q0 + lh * 8 + r;
      float v = o[nf][r] / rl[r];
      out[base + (size_t)qr * Dc + nf * 16 + fr] = f2bf(v);
    }
  }
}

// ---------------- host orchestration ----------------
extern "C" void kernel_launch(void* const* d_in, const int* in_sizes, int n_in,
                              void* d_out, int out_size, void* d_ws, size_t ws_size,
                              hipStream_t stream) {
  const float* x     = (const float*)d_in[0];
  const int*   mask  = (const int*)d_in[1];
  const float* wq    = (const float*)d_in[2];
  const float* bq    = (const float*)d_in[3];
  const float* wk    = (const float*)d_in[4];
  const float* bk    = (const float*)d_in[5];
  const float* wv    = (const float*)d_in[6];
  const float* bv    = (const float*)d_in[7];
  const float* wo    = (const float*)d_in[8];
  const float* bo    = (const float*)d_in[9];
  const float* w1    = (const float*)d_in[10];
  const float* b1    = (const float*)d_in[11];
  const float* w2    = (const float*)d_in[12];
  const float* b2    = (const float*)d_in[13];
  const float* ln1g  = (const float*)d_in[14];
  const float* ln1b  = (const float*)d_in[15];
  const float* ln2g  = (const float*)d_in[16];
  const float* ln2b  = (const float*)d_in[17];
  float* out = (float*)d_out;

  char* ws = (char*)d_ws;
  size_t off = 0;
  auto carve = [&](size_t bytes) -> void* {
    void* p = ws + off;
    off = (off + bytes + 255) & ~(size_t)255;
    return p;
  };
  u16*   h_bf   = (u16*)  carve((size_t)MT * Dc * 2);
  u16*   q_bf   = (u16*)  carve((size_t)MT * Dc * 2);
  u16*   k_bf   = (u16*)  carve((size_t)MT * Dc * 2);
  u16*   v_bf   = (u16*)  carve((size_t)MT * Dc * 2);
  u16*   at_bf  = (u16*)  carve((size_t)MT * Dc * 2);
  u16*   h2_bf  = (u16*)  carve((size_t)MT * Dc * 2);
  u16*   ff_bf  = (u16*)  carve((size_t)MT * DFFc * 2);
  float* x1     = (float*)carve((size_t)MT * Dc * 4);
  // pre-transposed bf16 weights: [N][K]
  u16*   wqt    = (u16*)  carve((size_t)Dc * Dc * 2);
  u16*   wkt    = (u16*)  carve((size_t)Dc * Dc * 2);
  u16*   wvt    = (u16*)  carve((size_t)Dc * Dc * 2);
  u16*   wot    = (u16*)  carve((size_t)Dc * Dc * 2);
  u16*   w1t    = (u16*)  carve((size_t)Dc * DFFc * 2);
  u16*   w2t    = (u16*)  carve((size_t)DFFc * Dc * 2);
  unsigned* amx = (unsigned*)carve(16);

  const int nDD = Dc * Dc, nDF = Dc * DFFc;
  dim3 blk(256);

  // 1) fake-quant scales for wq/wk/wv; convert+transpose all weights
  init_amax_kernel<<<1, 32, 0, stream>>>(amx);
  absmax_kernel<<<256, blk, 0, stream>>>(wq, nDD, amx + 0);
  absmax_kernel<<<256, blk, 0, stream>>>(wk, nDD, amx + 1);
  absmax_kernel<<<256, blk, 0, stream>>>(wv, nDD, amx + 2);
  quant_bf16_t_kernel<<<(nDD + 255) / 256, blk, 0, stream>>>(wq, wqt, Dc, Dc, amx + 0);
  quant_bf16_t_kernel<<<(nDD + 255) / 256, blk, 0, stream>>>(wk, wkt, Dc, Dc, amx + 1);
  quant_bf16_t_kernel<<<(nDD + 255) / 256, blk, 0, stream>>>(wv, wvt, Dc, Dc, amx + 2);
  cvt_bf16_t_kernel<<<(nDD + 255) / 256, blk, 0, stream>>>(wo, wot, Dc, Dc);
  cvt_bf16_t_kernel<<<(nDF + 255) / 256, blk, 0, stream>>>(w1, w1t, Dc, DFFc);   // -> [DFF][D]
  cvt_bf16_t_kernel<<<(nDF + 255) / 256, blk, 0, stream>>>(w2, w2t, DFFc, Dc);   // -> [D][DFF]

  // 2) LN1 -> h
  ln_bf16_kernel<<<MT, blk, 0, stream>>>(x, ln1g, ln1b, h_bf);

  // 3) QKV projections
  dim3 gD(Dc / BNt, MT / BMt);   // (8, 64)
  gemm_bf16_kernel<0><<<gD, blk, 0, stream>>>(h_bf, wqt, bq, nullptr, nullptr, q_bf, Dc, Dc);
  gemm_bf16_kernel<0><<<gD, blk, 0, stream>>>(h_bf, wkt, bk, nullptr, nullptr, k_bf, Dc, Dc);
  gemm_bf16_kernel<0><<<gD, blk, 0, stream>>>(h_bf, wvt, bv, nullptr, nullptr, v_bf, Dc, Dc);

  // 4) attention
  attention_kernel<<<Bc * Hc * (Sc / 128), blk, 0, stream>>>(q_bf, k_bf, v_bf, mask, at_bf);

  // 5) output projection + residual -> x1 (f32)
  gemm_bf16_kernel<2><<<gD, blk, 0, stream>>>(at_bf, wot, bo, x, x1, nullptr, Dc, Dc);

  // 6) LN2 -> h2
  ln_bf16_kernel<<<MT, blk, 0, stream>>>(x1, ln2g, ln2b, h2_bf);

  // 7) FFN
  dim3 gF(DFFc / BNt, MT / BMt); // (32, 64)
  gemm_bf16_kernel<1><<<gF, blk, 0, stream>>>(h2_bf, w1t, b1, nullptr, nullptr, ff_bf, DFFc, Dc);
  gemm_bf16_kernel<2><<<gD, blk, 0, stream>>>(ff_bf, w2t, b2, x1, out, nullptr, Dc, DFFc);

  (void)in_sizes; (void)n_in; (void)out_size; (void)ws_size;
}
